// Block_en_49469433315543
// MI455X (gfx1250) — matrix operand order repaired and checked by reading, not compiled, not run on hardware
//
#include <hip/hip_runtime.h>

// Problem constants (from setup_inputs): x [B=8, C=64, H=128, W=128] fp32.
#define B_    8
#define C_    64
#define H_    128
#define W_    128
#define HW_   (H_ * W_)
#define NPIX_ (B_ * HW_)   // 131072 pixels

typedef __attribute__((ext_vector_type(16))) __bf16        v16bf;
typedef __attribute__((ext_vector_type(8)))  float         v8f;
typedef __attribute__((ext_vector_type(4)))  unsigned int  u32x4;

union BF16Frag {
  v16bf          v;
  unsigned short h[16];
  unsigned int   w[8];
  u32x4          q[2];
};

__device__ __forceinline__ unsigned short f2bf(float f) {
  unsigned int u = __float_as_uint(f);
  u += 0x7FFFu + ((u >> 16) & 1u);     // round-to-nearest-even
  return (unsigned short)(u >> 16);
}

__device__ __forceinline__ float softplus_f(float x) {
  // stable softplus: max(x,0) + log1p(exp(-|x|))
  return fmaxf(x, 0.0f) + log1pf(__expf(-fabsf(x)));
}

// Fragment-order helper (A matrix, 16-bit, 16x32, contiguous k order used in this file):
//   lane = half*16 + nl ; element e = 2*ii + lo (ii = VGPR idx 0..7)
//   m  = nl
//   k  = kt*32 + half*16 + e
__device__ __forceinline__ int a_frag_k(int kt, int half, int e) {
  return kt * 32 + (half << 4) + e;
}

// ---------------------------------------------------------------------------
// Stage 0a: swizzle conv2 weights [co][ci][kh][kw] fp32 -> bf16 fragment-major
//   wbs[ ((tap*4 + ct)*2 + kt)*32 + lane ][16]  so each lane's A fragment is
//   2 contiguous 16B chunks (two global_load_b128, coalesced across the wave).
// ---------------------------------------------------------------------------
__global__ __launch_bounds__(256) void k_wconv(const float* __restrict__ w_conv2,
                                               unsigned short* __restrict__ wbs) {
  int i = blockIdx.x * 256 + threadIdx.x;   // over 9*64*64 = 36864
  if (i >= 9 * C_ * C_) return;
  int e    = i & 15;
  int lane = (i >> 4) & 31;
  int kt   = (i >> 9) & 1;
  int ct   = (i >> 10) & 3;
  int tap  = i >> 12;
  int nl = lane & 15, half = lane >> 4;
  int co = ct * 16 + nl;
  int ci = a_frag_k(kt, half, e);
  int kh = tap / 3, kw = tap - kh * 3;
  wbs[i] = f2bf(w_conv2[((co * C_ + ci) * 3 + kh) * 3 + kw]);
}

// Stage 0b: same swizzle for the 64x64 reduce weights (single "tap").
__global__ __launch_bounds__(256) void k_wred(const float* __restrict__ w_reduce,
                                              unsigned short* __restrict__ wrs) {
  int i = blockIdx.x * 256 + threadIdx.x;   // over 64*64 = 4096
  if (i >= C_ * C_) return;
  int e    = i & 15;
  int lane = (i >> 4) & 31;
  int kt   = (i >> 9) & 1;
  int ct   = (i >> 10) & 3;
  int nl = lane & 15, half = lane >> 4;
  int co = ct * 16 + nl;
  int ci = a_frag_k(kt, half, e);
  wrs[i] = f2bf(w_reduce[co * C_ + ci]);
}

// ---------------------------------------------------------------------------
// Stage 1: r[64, N] = W_reduce(64x64) . x[64, N] + b_reduce via WMMA bf16.
// Block = 256 threads (8 waves) -> 64 outC x 32 px; x converted fp32->bf16
// while staging into LDS [px][ci(+pad 72)] so each lane's B fragment is two
// ds_load_b128 (16 consecutive ci at fixed px).
// ---------------------------------------------------------------------------
__global__ __launch_bounds__(256) void k_reduce_wmma(const float* __restrict__ x,
                                                     const unsigned short* __restrict__ wrs,
                                                     const float* __restrict__ b_reduce,
                                                     float* __restrict__ r) {
  __shared__ __align__(16) unsigned short sx[32 * 72];

  int n0   = blockIdx.x * 32;        // 4096 blocks; 32 px never straddles batch
  int b    = n0 / HW_;
  int rem0 = n0 - b * HW_;
  int tid  = threadIdx.x;

  for (int idx = tid; idx < C_ * 32; idx += 256) {
    int c  = idx >> 5;
    int px = idx & 31;                // consecutive tid -> consecutive px (coalesced)
    float v = x[(size_t)(b * C_ + c) * HW_ + rem0 + px];
    sx[px * 72 + c] = f2bf(v);
  }
  __syncthreads();

  int lane = tid & 31, wave = tid >> 5;
  int co_base = (wave & 3) * 16;
  int px_base = (wave >> 2) * 16;
  int nl = lane & 15, half = lane >> 4;

  const u32x4* wq = (const u32x4*)wrs;
  v8f acc = {};
#pragma unroll
  for (int kt = 0; kt < 2; ++kt) {
    BF16Frag A, Bf;
    int fi = (((wave & 3) * 2 + kt) * 32 + lane) * 2;   // u32x4 units
    A.q[0] = wq[fi];
    A.q[1] = wq[fi + 1];
    const unsigned short* bp = &sx[(px_base + nl) * 72 + kt * 32 + half * 16];
    Bf.q[0] = *(const u32x4*)bp;
    Bf.q[1] = *(const u32x4*)(bp + 8);
    acc = __builtin_amdgcn_wmma_f32_16x16x32_bf16(false, A.v, false, Bf.v,
                                                  (short)0, acc, false, false);
  }

  int n = n0 + px_base + nl;
#pragma unroll
  for (int j = 0; j < 8; ++j) {
    int co = co_base + j + (half << 3);
    r[(size_t)co * NPIX_ + n] = acc[j] + b_reduce[co];
  }
}

// ---------------------------------------------------------------------------
// Stage 2: per-channel batch stats -> folded BN scale/shift
//   stats[c] = gamma*rsqrt(var+eps) ; stats[64+c] = beta - mean*stats[c]
// ---------------------------------------------------------------------------
__global__ __launch_bounds__(256) void k_stats(const float* __restrict__ r,
                                               const float* __restrict__ gamma,
                                               const float* __restrict__ beta,
                                               float* __restrict__ stats) {
  __shared__ float s1[256], s2[256];
  int c = blockIdx.x, tid = threadIdx.x;
  const float* rc = r + (size_t)c * NPIX_;
  float a = 0.0f, b = 0.0f;
  for (int i = tid; i < NPIX_; i += 256) {
    float v = rc[i];
    a += v;
    b = fmaf(v, v, b);
  }
  s1[tid] = a; s2[tid] = b;
  __syncthreads();
  for (int off = 128; off > 0; off >>= 1) {
    if (tid < off) { s1[tid] += s1[tid + off]; s2[tid] += s2[tid + off]; }
    __syncthreads();
  }
  if (tid == 0) {
    const float inv = 1.0f / (float)NPIX_;
    float mu   = s1[0] * inv;
    float var  = s2[0] * inv - mu * mu;
    float rstd = rsqrtf(var + 1e-5f);
    float sc   = gamma[c] * rstd;
    stats[c]      = sc;
    stats[C_ + c] = beta[c] - mu * sc;
  }
}

// ---------------------------------------------------------------------------
// Stage 3: BN+ReLU+span -> per-pixel 9-tap kernel; involution apply over 3x3
//          patches; softplus; store z as bf16 NCHW. (Per-pixel diagonal op,
//          stays on VALU fp32.)
// ---------------------------------------------------------------------------
__global__ __launch_bounds__(256) void k_invol(const float* __restrict__ x,
                                               const float* __restrict__ r,
                                               const float* __restrict__ stats,
                                               const float* __restrict__ w_span,
                                               const float* __restrict__ b_span,
                                               unsigned short* __restrict__ z) {
  int n   = blockIdx.x * 256 + threadIdx.x;
  int b   = n / HW_;
  int rem = n - b * HW_;
  int h   = rem / W_;
  int w   = rem - h * W_;

  float kern[9];
#pragma unroll
  for (int k = 0; k < 9; ++k) kern[k] = b_span[k];
  for (int c = 0; c < C_; ++c) {
    float rn = fmaf(stats[c], r[(size_t)c * NPIX_ + n], stats[C_ + c]);
    rn = fmaxf(rn, 0.0f);                         // ReLU
#pragma unroll
    for (int k = 0; k < 9; ++k) kern[k] = fmaf(w_span[k * C_ + c], rn, kern[k]);
  }

  const float* xb = x + (size_t)b * C_ * HW_;
  for (int c = 0; c < C_; ++c) {
    const float* xc = xb + c * HW_;
    float s = 0.0f;
#pragma unroll
    for (int ky = 0; ky < 3; ++ky) {
      int gh = h + ky - 1;
      if (gh < 0 || gh >= H_) continue;
#pragma unroll
      for (int kx = 0; kx < 3; ++kx) {
        int gw = w + kx - 1;
        if (gw < 0 || gw >= W_) continue;
        s = fmaf(kern[ky * 3 + kx], xc[gh * W_ + gw], s);
      }
    }
    z[(size_t)(b * C_ + c) * HW_ + rem] = f2bf(softplus_f(s));
  }
}

// ---------------------------------------------------------------------------
// Stage 4: 3x3 conv 64->64 as implicit GEMM on v_wmma_f32_16x16x32_bf16.
// Block = 256 threads (8 waves) -> 64 outC x 32 px tile at (b, h, w0..w0+31).
// LDS slab [row][x][ci(+pad 72)]: B fragment = two ds_load_b128 per K-tile;
// A fragment = two coalesced global_load_b128 from the swizzled weights.
// ---------------------------------------------------------------------------
__global__ __launch_bounds__(256) void k_conv_wmma(const unsigned short* __restrict__ z,
                                                   const unsigned short* __restrict__ wbs,
                                                   const float* __restrict__ b_conv2,
                                                   float* __restrict__ out) {
  __shared__ __align__(16) unsigned short sl[3 * 34 * 72];   // 14688 B

  int wt = blockIdx.x & 3;                        // W/32 tiles
  int h  = (blockIdx.x >> 2) & (H_ - 1);
  int b  = blockIdx.x >> 9;
  int w0 = wt * 32;
  int tid = threadIdx.x;

  // Stage slab: sl[(row*34 + xx)*72 + c], xx maps to gw = w0 + xx - 1.
  // Consecutive tid -> consecutive xx: coalesced global reads; LDS write
  // stride 36 dwords -> conflict-free.
  for (int idx = tid; idx < C_ * 3 * 34; idx += 256) {
    int c   = idx / 102;
    int rm  = idx - c * 102;
    int row = rm / 34;
    int xx  = rm - row * 34;
    int gh  = h + row - 1;
    int gw  = w0 + xx - 1;
    unsigned short v = 0;
    if (gh >= 0 && gh < H_ && gw >= 0 && gw < W_)
      v = z[(size_t)(b * C_ + c) * HW_ + gh * W_ + gw];
    sl[(row * 34 + xx) * 72 + c] = v;
  }
  __syncthreads();

  int lane = tid & 31;
  int wave = tid >> 5;
  int co_base = (wave & 3) * 16;
  int px_base = (wave >> 2) * 16;
  int nl   = lane & 15;
  int half = lane >> 4;

  const u32x4* wq = (const u32x4*)wbs;
  v8f acc = {};

  for (int tap = 0; tap < 9; ++tap) {
    int row  = tap / 3;                           // dy + 1
    int xcol = px_base + nl + (tap % 3);          // (px + dx) + 1 halo offset
    const unsigned short* bp0 = &sl[(row * 34 + xcol) * 72];
#pragma unroll
    for (int kt = 0; kt < 2; ++kt) {
      BF16Frag A, Bf;
      int fi = (((tap * 4 + (wave & 3)) * 2 + kt) * 32 + lane) * 2;  // u32x4 units
      A.q[0] = wq[fi];
      A.q[1] = wq[fi + 1];
      const unsigned short* bp = bp0 + kt * 32 + half * 16;   // 16 consecutive ci
      Bf.q[0] = *(const u32x4*)bp;
      Bf.q[1] = *(const u32x4*)(bp + 8);
      acc = __builtin_amdgcn_wmma_f32_16x16x32_bf16(false, A.v, false, Bf.v,
                                                    (short)0, acc, false, false);
    }
  }

  // D layout: VGPR j, lanes 0-15 -> M = j, lanes 16-31 -> M = j + 8; N = nl
  int pix = w0 + px_base + nl;
#pragma unroll
  for (int j = 0; j < 8; ++j) {
    int co = co_base + j + (half << 3);
    float v = acc[j] + b_conv2[co];
    out[((size_t)(b * C_ + co) * H_ + h) * W_ + pix] = softplus_f(v);
  }
}

// ---------------------------------------------------------------------------
extern "C" void kernel_launch(void* const* d_in, const int* in_sizes, int n_in,
                              void* d_out, int out_size, void* d_ws, size_t ws_size,
                              hipStream_t stream) {
  (void)in_sizes; (void)n_in; (void)out_size; (void)ws_size;

  const float* x        = (const float*)d_in[0];
  const float* w_reduce = (const float*)d_in[1];
  const float* b_reduce = (const float*)d_in[2];
  const float* bn_gamma = (const float*)d_in[3];
  const float* bn_beta  = (const float*)d_in[4];
  const float* w_span   = (const float*)d_in[5];
  const float* b_span   = (const float*)d_in[6];
  const float* w_conv2  = (const float*)d_in[7];
  const float* b_conv2  = (const float*)d_in[8];
  float* out = (float*)d_out;

  char* ws = (char*)d_ws;
  float*          r     = (float*)ws;                             // 33554432 B
  unsigned short* z     = (unsigned short*)(ws + 33554432);       // 16777216 B
  unsigned short* wbs   = (unsigned short*)(ws + 50331648);       //    73728 B
  unsigned short* wrs   = (unsigned short*)(ws + 50405376);       //     8192 B
  float*          stats = (float*)(ws + 50413568);                //      512 B

  k_wconv<<<dim3(144), dim3(256), 0, stream>>>(w_conv2, wbs);
  k_wred <<<dim3(16),  dim3(256), 0, stream>>>(w_reduce, wrs);
  k_reduce_wmma<<<dim3(NPIX_ / 32), dim3(256), 0, stream>>>(x, wrs, b_reduce, r);
  k_stats<<<dim3(C_), dim3(256), 0, stream>>>(r, bn_gamma, bn_beta, stats);
  k_invol<<<dim3(NPIX_ / 256), dim3(256), 0, stream>>>(x, r, stats, w_span, b_span, z);
  k_conv_wmma<<<dim3(B_ * H_ * (W_ / 32)), dim3(256), 0, stream>>>(z, wbs, b_conv2, out);
}
